// GuardGCN_13176959664522
// MI455X (gfx1250) — compile-verified, matrix operand order checked by reading, not run on hardware
//
#include <hip/hip_runtime.h>
#include <hip/hip_bf16.h>
#include <math.h>

typedef __attribute__((ext_vector_type(16))) _Float16 v16h;
typedef __attribute__((ext_vector_type(8)))  float    v8f;

#define NFEAT 128
#define NHID  64
#define NCLASS 40

// ---------------------------------------------------------------------------
// Per-node inverse (safe) L2 norm: one wave32 per node, float4 loads.
// ---------------------------------------------------------------------------
__global__ void k_norm(const float* __restrict__ F, int nf, float* __restrict__ inv, int N) {
    int node = blockIdx.x * (blockDim.x / 32) + (threadIdx.x / 32);
    int lane = threadIdx.x & 31;
    if (node >= N) return;
    const float* row = F + (size_t)node * nf;
    float acc = 0.f;
    for (int c = lane * 4; c + 3 < nf; c += 128) {
        float4 v = *(const float4*)(row + c);
        acc += v.x * v.x + v.y * v.y + v.z * v.z + v.w * v.w;
    }
    for (int off = 16; off > 0; off >>= 1) acc += __shfl_xor(acc, off, 32);
    if (lane == 0) {
        float nrm = sqrtf(acc);
        float safe = (nrm == 0.f) ? 1.f : nrm;
        inv[node] = 1.f / safe;
    }
}

// ---------------------------------------------------------------------------
// Per-edge cosine similarity, threshold at 0.1, optional mask, and
// rowsum[src] += sim (L1 segment sum; sim >= 0 after threshold).
// One wave32 per edge, float4 feature gathers (rows stay resident in L2).
// ---------------------------------------------------------------------------
__global__ void k_edge_sim(const float* __restrict__ F, int nf,
                           const int* __restrict__ src, const int* __restrict__ dst,
                           const float* __restrict__ inv, const float* __restrict__ mask,
                           float* __restrict__ sim, float* __restrict__ rowsum, int E) {
    int e = blockIdx.x * (blockDim.x / 32) + (threadIdx.x / 32);
    int lane = threadIdx.x & 31;
    if (e >= E) return;
    int s = src[e], d = dst[e];
    const float* fs = F + (size_t)s * nf;
    const float* fd = F + (size_t)d * nf;
    float acc = 0.f;
    for (int c = lane * 4; c + 3 < nf; c += 128) {
        float4 a = *(const float4*)(fs + c);
        float4 b = *(const float4*)(fd + c);
        acc += a.x * b.x + a.y * b.y + a.z * b.z + a.w * b.w;
    }
    for (int off = 16; off > 0; off >>= 1) acc += __shfl_xor(acc, off, 32);
    if (lane == 0) {
        float v = acc * inv[s] * inv[d];
        if (v < 0.1f) v = 0.f;
        if (mask != nullptr && !(mask[e] > 0.f)) v = 0.f;
        sim[e] = v;
        if (v != 0.f) atomicAdd(rowsum + s, v);
    }
}

// ---------------------------------------------------------------------------
// L1 normalize, learnable drop (sigmoid([a, a_rev]@Wd + bd) > 0.5 <=> z > 0),
// w_e = exp(a) for surviving a>0, deg[src] += (a != 0).
// ---------------------------------------------------------------------------
__global__ void k_drop(const float* __restrict__ sim, const int* __restrict__ src,
                       const int* __restrict__ rev, const float* __restrict__ rowsum,
                       const float* __restrict__ Wd, const float* __restrict__ bd,
                       float* __restrict__ w_e, float* __restrict__ deg, int E) {
    int e = blockIdx.x * blockDim.x + threadIdx.x;
    if (e >= E) return;
    float rs = rowsum[src[e]];  rs = (rs == 0.f) ? 1.f : rs;
    float a  = sim[e] / rs;
    int   re = rev[e];
    float rs2 = rowsum[src[re]]; rs2 = (rs2 == 0.f) ? 1.f : rs2;
    float ar  = sim[re] / rs2;
    float z = a * Wd[0] + ar * Wd[1] + bd[0];
    float av = (z > 0.f) ? a : 0.f;
    w_e[e] = (av > 0.f) ? expf(av) : 0.f;
    if (av != 0.f) atomicAdd(deg + src[e], 1.f);
}

// w_s[i] = exp(1/(deg[i]+1))
__global__ void k_selfw(const float* __restrict__ deg, float* __restrict__ w_s, int N) {
    int i = blockIdx.x * blockDim.x + threadIdx.x;
    if (i >= N) return;
    w_s[i] = expf(1.f / (deg[i] + 1.f));
}

// ---------------------------------------------------------------------------
// WMMA GEMM: C[M x NCOLS] = A[M x KDIM] * W[KDIM x NCOLS], f32 in/out,
// f16 multiply via v_wmma_f32_16x16x32_f16, f32 accumulate.
// Block = 128 threads = 4 waves; one 16-row M-tile per wave.
//
// A fragment (16-bit 16x32, per ISA 7.12.2): lane L<16 -> rows 0..15, K 0..7
// then 16..23; L>=16 -> K 8..15 then 24..31 (within the k-step). B mirrors it.
// B is staged in LDS PRE-SWIZZLED into fragment order: Bs[ktile][nt][lane][16]
// halves contiguous and 32B-aligned, so each wave reads a whole v16h fragment
// with two ds_load_b128 instead of 16 ds_load_u16.
// C/D layout: lane = column, VGPR r = row r + 8*(lane>=16).
// ---------------------------------------------------------------------------
template <int KDIM, int NT, int NCOLS>
__global__ void k_wmma_gemm(const float* __restrict__ A, const float* __restrict__ W,
                            float* __restrict__ C, int MT) {
    constexpr int KT = KDIM / 32;                       // k-steps
    alignas(32) __shared__ _Float16 Bs[KT][NT][32][16]; // fragment-order B

    // cooperative swizzle fill: element (kt, t, lane, idx) holds
    // W[kt*32 + 8*(lane>=16) + (idx<8 ? idx : idx+8)][t*16 + (lane&15)]
    for (int i = threadIdx.x; i < KT * NT * 32 * 16; i += blockDim.x) {
        int idx  = i & 15;
        int lane = (i >> 4) & 31;
        int t    = (i >> 9) % NT;
        int kt   = i / (16 * 32 * NT);
        int kk   = kt * 32 + ((lane >> 4) * 8) + ((idx < 8) ? idx : (idx + 8));
        int col  = t * 16 + (lane & 15);
        Bs[kt][t][lane][idx] = (col < NCOLS) ? (_Float16)W[kk * NCOLS + col]
                                             : (_Float16)0.f;
    }
    __syncthreads();

    int wave  = threadIdx.x / 32;
    int lane  = threadIdx.x & 31;
    int mtile = blockIdx.x * (blockDim.x / 32) + wave;
    if (mtile >= MT) return;

    int base = (lane >> 4) * 8;   // 0: K 0..7/16..23   8: K 8..15/24..31
    int l16  = lane & 15;
    int m    = mtile * 16 + l16;
    const float* arow = A + (size_t)m * KDIM;

    v8f acc[NT] = {};
#pragma unroll
    for (int kt = 0; kt < KT; ++kt) {
        int k0 = kt * 32;
        if (kt + 1 < KT) __builtin_prefetch(arow + k0 + 32, 0, 1);
        // A fragment: two contiguous runs of 8 floats -> 4x b128 loads + cvt
        float4 p0 = *(const float4*)(arow + k0 + base);
        float4 p1 = *(const float4*)(arow + k0 + base + 4);
        float4 p2 = *(const float4*)(arow + k0 + base + 16);
        float4 p3 = *(const float4*)(arow + k0 + base + 20);
        v16h a;
        a[0]  = (_Float16)p0.x; a[1]  = (_Float16)p0.y; a[2]  = (_Float16)p0.z; a[3]  = (_Float16)p0.w;
        a[4]  = (_Float16)p1.x; a[5]  = (_Float16)p1.y; a[6]  = (_Float16)p1.z; a[7]  = (_Float16)p1.w;
        a[8]  = (_Float16)p2.x; a[9]  = (_Float16)p2.y; a[10] = (_Float16)p2.z; a[11] = (_Float16)p2.w;
        a[12] = (_Float16)p3.x; a[13] = (_Float16)p3.y; a[14] = (_Float16)p3.z; a[15] = (_Float16)p3.w;
#pragma unroll
        for (int t = 0; t < NT; ++t) {
            v16h b = *reinterpret_cast<const v16h*>(&Bs[kt][t][lane][0]);
            acc[t] = __builtin_amdgcn_wmma_f32_16x16x32_f16(
                false, a, false, b, (short)0, acc[t], false, false);
        }
    }
#pragma unroll
    for (int t = 0; t < NT; ++t) {
        int n = t * 16 + l16;
        if (n < NCOLS) {
#pragma unroll
            for (int r = 0; r < 8; ++r) {
                int row = mtile * 16 + r + base;   // base = 8*(lane>=16)
                C[(size_t)row * NCOLS + n] = acc[t][r];
            }
        }
    }
}

// degw[i] = w_s[i] + 1 (the two self-loops of gcn_norm), then edge atomics add.
__global__ void k_deg_init(float* __restrict__ degw, const float* __restrict__ w_s, int N) {
    int i = blockIdx.x * blockDim.x + threadIdx.x;
    if (i >= N) return;
    degw[i] = w_s[i] + 1.f;
}

__global__ void k_deg_edge(float* __restrict__ degw, const int* __restrict__ dst,
                           const float* __restrict__ w_e, int E) {
    int e = blockIdx.x * blockDim.x + threadIdx.x;
    if (e >= E) return;
    float w = w_e[e];
    if (w != 0.f) atomicAdd(degw + dst[e], w);
}

// degw -> D^{-1/2} in place
__global__ void k_dis(float* __restrict__ degw, int N) {
    int i = blockIdx.x * blockDim.x + threadIdx.x;
    if (i >= N) return;
    float d = degw[i];
    degw[i] = (d > 0.f) ? rsqrtf(d) : 0.f;
}

// out[i][c] = bias[c] + dis[i]^2 * (w_s[i] + 1) * hlin[i][c]   (self loops)
__global__ void k_init_out(float* __restrict__ out, const float* __restrict__ bias,
                           const float* __restrict__ dis, const float* __restrict__ w_s,
                           const float* __restrict__ hlin, int N, int Cc) {
    long long idx = (long long)blockIdx.x * blockDim.x + threadIdx.x;
    if (idx >= (long long)N * Cc) return;
    int i = (int)(idx / Cc), c = (int)(idx % Cc);
    float di = dis[i];
    out[idx] = bias[c] + di * di * (w_s[i] + 1.f) * hlin[idx];
}

// out[dst] += dis[src]*w*dis[dst] * hlin[src]   (wave32 per edge, lane = cols)
__global__ void k_agg_edge(float* __restrict__ out, const int* __restrict__ src,
                           const int* __restrict__ dst, const float* __restrict__ w_e,
                           const float* __restrict__ dis, const float* __restrict__ hlin,
                           int E, int Cc) {
    int e = blockIdx.x * (blockDim.x / 32) + (threadIdx.x / 32);
    int lane = threadIdx.x & 31;
    if (e >= E) return;
    float w = w_e[e];
    if (w == 0.f) return;
    int s = src[e], d = dst[e];
    float scale = dis[s] * w * dis[d];
    const float* hs = hlin + (size_t)s * Cc;
    float* od = out + (size_t)d * Cc;
    for (int c = lane; c < Cc; c += 32) atomicAdd(od + c, scale * hs[c]);
}

__global__ void k_relu(float* __restrict__ h, long long n) {
    long long i = (long long)blockIdx.x * blockDim.x + threadIdx.x;
    if (i >= n) return;
    float v = h[i];
    h[i] = (v > 0.f) ? v : 0.f;
}

// In-place log_softmax over rows of width Cc (wave32 per row).
__global__ void k_logsoftmax(float* __restrict__ out, int N, int Cc) {
    int row = blockIdx.x * (blockDim.x / 32) + (threadIdx.x / 32);
    int lane = threadIdx.x & 31;
    if (row >= N) return;
    float* r = out + (size_t)row * Cc;
    float m = -INFINITY;
    for (int c = lane; c < Cc; c += 32) m = fmaxf(m, r[c]);
    for (int off = 16; off > 0; off >>= 1) m = fmaxf(m, __shfl_xor(m, off, 32));
    float s = 0.f;
    for (int c = lane; c < Cc; c += 32) s += expf(r[c] - m);
    for (int off = 16; off > 0; off >>= 1) s += __shfl_xor(s, off, 32);
    float lse = m + logf(s);
    for (int c = lane; c < Cc; c += 32) r[c] = r[c] - lse;
}

// ---------------------------------------------------------------------------
extern "C" void kernel_launch(void* const* d_in, const int* in_sizes, int n_in,
                              void* d_out, int out_size, void* d_ws, size_t ws_size,
                              hipStream_t stream) {
    const float* x   = (const float*)d_in[0];
    const int*   src = (const int*)  d_in[1];
    const int*   dst = (const int*)  d_in[2];
    const int*   rev = (const int*)  d_in[3];
    const float* W1  = (const float*)d_in[4];
    const float* b1  = (const float*)d_in[5];
    const float* W2  = (const float*)d_in[6];
    const float* b2  = (const float*)d_in[7];
    const float* Wd  = (const float*)d_in[8];
    const float* bd  = (const float*)d_in[9];
    float* out = (float*)d_out;

    const int N = in_sizes[0] / NFEAT;   // 100000
    const int E = in_sizes[1];           // 1000000

    // workspace carve-up (floats); all offsets are multiples of 16 bytes
    float* W = (float*)d_ws;
    float* inv    = W;              W += N;
    float* sim    = W;              W += E;
    float* rowsum = W;              W += N;
    float* w_e1   = W;              W += E;
    float* w_s1   = W;              W += N;
    float* deg    = W;              W += N;
    float* degw   = W;              W += N;
    float* h_lin  = W;              W += (size_t)N * NHID;  // reused as N x 40 in layer 2
    float* h      = W;              W += (size_t)N * NHID;
    float* w_e2   = W;              W += E;
    float* w_s2   = W;              W += N;
    (void)ws_size; (void)n_in; (void)out_size;

    const int TB = 256;
    dim3 blk(TB);
    int wavesPerBlk = TB / 32;
    dim3 gNodeWave((N + wavesPerBlk - 1) / wavesPerBlk);
    dim3 gEdgeWave((E + wavesPerBlk - 1) / wavesPerBlk);
    dim3 gEdge((E + TB - 1) / TB);
    dim3 gNode((N + TB - 1) / TB);

    const int MT = (N + 15) / 16;                 // 6250 M-tiles
    dim3 gGemm((MT + 3) / 4);                     // 4 waves / block
    dim3 bGemm(128);

    // ---------------- pass 1: attention on raw features ----------------
    hipMemsetAsync(rowsum, 0, N * sizeof(float), stream);
    hipMemsetAsync(deg,    0, N * sizeof(float), stream);
    k_norm<<<gNodeWave, blk, 0, stream>>>(x, NFEAT, inv, N);
    k_edge_sim<<<gEdgeWave, blk, 0, stream>>>(x, NFEAT, src, dst, inv, nullptr, sim, rowsum, E);
    k_drop<<<gEdge, blk, 0, stream>>>(sim, src, rev, rowsum, Wd, bd, w_e1, deg, E);
    k_selfw<<<gNode, blk, 0, stream>>>(deg, w_s1, N);

    // ---------------- layer 1: GCNConv + ReLU ----------------
    k_wmma_gemm<NFEAT, 4, NHID><<<gGemm, bGemm, 0, stream>>>(x, W1, h_lin, MT);
    k_deg_init<<<gNode, blk, 0, stream>>>(degw, w_s1, N);
    k_deg_edge<<<gEdge, blk, 0, stream>>>(degw, dst, w_e1, E);
    k_dis<<<gNode, blk, 0, stream>>>(degw, N);
    {
        long long nel = (long long)N * NHID;
        dim3 gEl((unsigned)((nel + TB - 1) / TB));
        k_init_out<<<gEl, blk, 0, stream>>>(h, b1, degw, w_s1, h_lin, N, NHID);
        k_agg_edge<<<gEdgeWave, blk, 0, stream>>>(h, src, dst, w_e1, degw, h_lin, E, NHID);
        k_relu<<<gEl, blk, 0, stream>>>(h, nel);
    }

    // ---------------- pass 2: attention on hidden features ----------------
    hipMemsetAsync(rowsum, 0, N * sizeof(float), stream);
    hipMemsetAsync(deg,    0, N * sizeof(float), stream);
    k_norm<<<gNodeWave, blk, 0, stream>>>(h, NHID, inv, N);
    k_edge_sim<<<gEdgeWave, blk, 0, stream>>>(h, NHID, src, dst, inv, w_e1, sim, rowsum, E);
    k_drop<<<gEdge, blk, 0, stream>>>(sim, src, rev, rowsum, Wd, bd, w_e2, deg, E);
    k_selfw<<<gNode, blk, 0, stream>>>(deg, w_s2, N);

    // ---------------- layer 2: GCNConv + log_softmax ----------------
    k_wmma_gemm<NHID, 3, NCLASS><<<gGemm, bGemm, 0, stream>>>(h, W2, h_lin, MT);
    k_deg_init<<<gNode, blk, 0, stream>>>(degw, w_s2, N);
    k_deg_edge<<<gEdge, blk, 0, stream>>>(degw, dst, w_e2, E);
    k_dis<<<gNode, blk, 0, stream>>>(degw, N);
    {
        long long nel = (long long)N * NCLASS;
        dim3 gEl((unsigned)((nel + TB - 1) / TB));
        k_init_out<<<gEl, blk, 0, stream>>>(out, b2, degw, w_s2, h_lin, N, NCLASS);
        k_agg_edge<<<gEdgeWave, blk, 0, stream>>>(out, src, dst, w_e2, degw, h_lin, E, NCLASS);
        k_logsoftmax<<<gNodeWave, blk, 0, stream>>>(out, N, NCLASS);
    }
}
